// GeodesicAttention_83923660964340
// MI455X (gfx1250) — matrix operand order repaired and checked by reading, not compiled, not run on hardware
//
#include <hip/hip_runtime.h>

// GeodesicAttention for MI455X (gfx1250, wave32).
//
// Linearization: Ad and the tanh argument are affine in (q_i, k_j), so all
// RANK-space projections are precomputed per token. ||diff||^2 expands to
// ||q||^2+||k||^2-2 q.k; the Gram matrix and attn@V both run on
// v_wmma_f32_16x16x32_f16. Scores are clipped to [-50,0] by the reference,
// so softmax is a single pass (exp never overflows/underflows the f32 sum).
// Working set ~5MB -> fully L2 resident (192MB L2); compute ~3 GFLOP.
// Constant folding: tanh arg = (0.5*qW+bm) + (0.5*kW); dist Gram term =
// fma(-2*lam, g, lam*qn) + (lam*kn) -- all per-side factors precomputed.

typedef _Float16 v16h __attribute__((ext_vector_type(16)));
typedef float    v8f  __attribute__((ext_vector_type(8)));

constexpr int Bc = 2, Sc = 1024, DIMc = 128, Hc = 4, HDc = 32, RKc = 16;
constexpr float INV_SQRT_HD = 0.17677669529663687f;   // 1/sqrt(32)
constexpr float LOG2E       = 1.4426950408889634f;

__device__ __forceinline__ float fast_exp(float x) {
  return __builtin_amdgcn_exp2f(x * LOG2E);           // v_exp_f32
}
__device__ __forceinline__ float fast_tanh(float x) {
  // tanh(x) = 1 - 2/(e^{2x}+1)
  float e = __builtin_amdgcn_exp2f((2.0f * LOG2E) * x);
  return 1.0f - 2.0f * __builtin_amdgcn_rcpf(e + 1.0f);
}

// ---------------------------------------------------------------------------
// Kernel 1: q,k,v = x@W.T+b ; per-head precomputes:
//   qA=q@A.T, kA=k@A.T, qW=q@Wm.T, kW=k@Wm.T, qn=|q|^2, kn=|k|^2
//   f16 repacks: Qh,Kh row-major [bh][s][hd]; Vt transposed [bh][hd][s]
// 8 tokens per block of 128 threads (amortizes weight reads).
// ---------------------------------------------------------------------------
__global__ __launch_bounds__(128) void proj_kernel(
    const float* __restrict__ x,
    const float* __restrict__ Wq, const float* __restrict__ bq,
    const float* __restrict__ Wk, const float* __restrict__ bk,
    const float* __restrict__ Wv, const float* __restrict__ bv,
    const float* __restrict__ Amat, const float* __restrict__ Wm,
    _Float16* __restrict__ Qh, _Float16* __restrict__ Kh, _Float16* __restrict__ Vt,
    float* __restrict__ qA, float* __restrict__ kA,
    float* __restrict__ qW, float* __restrict__ kW,
    float* __restrict__ qn, float* __restrict__ kn)
{
  __shared__ float xs[8][DIMc];
  __shared__ float qs[8][DIMc];
  __shared__ float ks[8][DIMc];

  const int tid = threadIdx.x;
  const int t0  = blockIdx.x * 8;          // flat token base (b*S+s)

  for (int idx = tid; idx < 8 * DIMc; idx += 128)
    xs[idx >> 7][idx & 127] = x[(size_t)t0 * DIMc + idx];
  __syncthreads();

  const int d = tid;                        // output feature dim 0..127
  float aq[8], ak[8], av[8];
#pragma unroll
  for (int t = 0; t < 8; ++t) { aq[t] = bq[d]; ak[t] = bk[d]; av[t] = bv[d]; }
  for (int c = 0; c < DIMc; ++c) {
    float wq = Wq[d * DIMc + c], wk = Wk[d * DIMc + c], wv = Wv[d * DIMc + c];
#pragma unroll
    for (int t = 0; t < 8; ++t) {
      float xv = xs[t][c];
      aq[t] += xv * wq; ak[t] += xv * wk; av[t] += xv * wv;
    }
  }

  const int h = d >> 5, hd = d & 31;
#pragma unroll
  for (int t = 0; t < 8; ++t) {
    int tok = t0 + t, b = tok >> 10, s = tok & 1023;
    int bh = b * Hc + h;
    qs[t][d] = aq[t]; ks[t][d] = ak[t];
    Qh[((size_t)bh * Sc + s) * HDc + hd] = (_Float16)aq[t];
    Kh[((size_t)bh * Sc + s) * HDc + hd] = (_Float16)ak[t];
    Vt[((size_t)bh * HDc + hd) * Sc + s] = (_Float16)av[t];
  }
  __syncthreads();

  // RANK-space projections: 4 arrays x (4 heads x 16 ranks) = 256 outputs/token
  for (int phase = 0; phase < 2; ++phase) {
    int oid = tid + phase * 128;            // 0..255
    int arr = oid >> 6;                     // 0:qA 1:kA 2:qW 3:kW
    int hh  = (oid >> 4) & 3;
    int r   = oid & 15;
    const float* wrow = ((arr < 2) ? Amat : Wm) + (hh * RKc + r) * HDc;
#pragma unroll
    for (int t = 0; t < 8; ++t) {
      const float* src = ((arr & 1) ? ks[t] : qs[t]) + hh * HDc;
      float acc = 0.0f;
#pragma unroll
      for (int e = 0; e < HDc; ++e) acc += src[e] * wrow[e];
      int tok = t0 + t, b = tok >> 10, s = tok & 1023;
      int bh  = b * Hc + hh;
      float* dst = (arr == 0) ? qA : (arr == 1) ? kA : (arr == 2) ? qW : kW;
      dst[((size_t)bh * Sc + s) * RKc + r] = acc;
    }
  }

  // Norms: 64 threads cover 8 tokens x 4 heads x {q,k}
  if (tid < 64) {
    int side = tid >> 5, idx = tid & 31, t = idx >> 2, hh = idx & 3;
    const float* src = (side ? ks[t] : qs[t]) + hh * HDc;
    float acc = 0.0f;
#pragma unroll
    for (int e = 0; e < HDc; ++e) acc += src[e] * src[e];
    int tok = t0 + t, b = tok >> 10, s = tok & 1023;
    int bh  = b * Hc + hh;
    (side ? kn : qn)[(size_t)bh * Sc + s] = acc;
  }
}

// ---------------------------------------------------------------------------
// Kernel 2: fused attention. Grid = B*H*16 = 128 WGs; 4 waves/WG; each wave
// owns a 16-row query tile of one (b,h). Per 32-column j chunk:
//   2x WMMA Gram (q.k), LDS transpose of G into the A-operand layout,
//   per-pair dist/tanh/exp in f32 (q-side rank data in per-lane registers,
//   k-side staged in LDS with constants pre-folded), then 2x WMMA for attn@V
//   (weights land already in A-layout; V pre-transposed so B is contiguous).
// ---------------------------------------------------------------------------
__global__ __launch_bounds__(128) void attn_kernel(
    const _Float16* __restrict__ Qh, const _Float16* __restrict__ Kh,
    const _Float16* __restrict__ Vt,
    const float* __restrict__ qA, const float* __restrict__ kA,
    const float* __restrict__ qW, const float* __restrict__ kW,
    const float* __restrict__ qn, const float* __restrict__ kn,
    const float* __restrict__ log_lam, const float* __restrict__ bm,
    float* __restrict__ Obuf)
{
  __shared__ float skA[32 * RKc];
  __shared__ float skW[32 * RKc];     // staged as 0.5*kW
  __shared__ float skn[32];           // staged as lam*kn
  __shared__ float sG[4][16 * 40];    // per-wave Gram transpose scratch (pad 40)

  const int tid  = threadIdx.x;
  const int wave = tid >> 5, lane = tid & 31;
  const int lm   = lane & 15, hi = lane >> 4;
  const int wg   = blockIdx.x;
  const int bh   = wg >> 4;                  // 16 WGs per (b,h)
  const int b    = bh >> 2, h = bh & 3;
  const int i0   = (wg & 15) * 64 + wave * 16;
  const int row  = i0 + lm;                  // this lane's query row

  const float lam   = fast_exp(log_lam[h]);
  const float lqn   = lam * qn[(size_t)bh * Sc + row];   // lam*|q_i|^2
  const float m2lam = -2.0f * lam;

  float qa_r[16], qwb[16];            // qA row; 0.5*qW row + bm (folded)
  {
    const float* qArow = qA + ((size_t)bh * Sc + row) * RKc;
    const float* qWrow = qW + ((size_t)bh * Sc + row) * RKc;
#pragma unroll
    for (int r = 0; r < 16; ++r) {
      qa_r[r] = qArow[r];
      qwb[r]  = 0.5f * qWrow[r] + bm[h * RKc + r];
    }
  }

  // A operand: Q tile 16x32 f16: lane holds row m=l&15,
  // K run [hi*16, hi*16+16) (contiguous, matches B).
  const int ko = hi * 16;
  v16h qa16;
  {
    const _Float16* qhrow = Qh + ((size_t)bh * Sc + row) * HDc;
#pragma unroll
    for (int e = 0; e < 8; ++e)  qa16[e] = qhrow[ko + e];
#pragma unroll
    for (int e = 8; e < 16; ++e) qa16[e] = qhrow[ko + e];
  }

  v8f o0 = {}, o1 = {};
  float rsum = 0.0f;
  float* gs = sG[wave];

  const float* kAbase = kA + (size_t)bh * Sc * RKc;
  const float* kWbase = kW + (size_t)bh * Sc * RKc;
  const float* knbase = kn + (size_t)bh * Sc;

  for (int j0 = 0; j0 < Sc; j0 += 32) {
    __syncthreads();
    // Stage k-side rank data for this 32-column chunk (constants folded in).
    {
      const float* kAsrc = kAbase + (size_t)j0 * RKc;
      const float* kWsrc = kWbase + (size_t)j0 * RKc;
      for (int idx = tid; idx < 32 * RKc; idx += 128) {
        skA[idx] = kAsrc[idx];
        skW[idx] = 0.5f * kWsrc[idx];
      }
      if (tid < 32) skn[tid] = lam * knbase[j0 + tid];
      // Prefetch next chunk's staging data into cache (global_prefetch_b8).
      if (j0 + 32 < Sc) {
        __builtin_prefetch(kAsrc + 32 * RKc + tid * 4, 0, 3);
        __builtin_prefetch(kWsrc + 32 * RKc + tid * 4, 0, 3);
      }
    }
    __syncthreads();

    // Gram tiles: D[m][n] = sum_hd Q[i0+m][hd] * K[j0+n][hd]
    // B operand: lane holds col n=l&15, K run [hi*16, hi*16+16) -> contiguous.
    const _Float16* kp0 = Kh + ((size_t)bh * Sc + j0 + lm) * HDc + hi * 16;
    v16h kb0 = *(const v16h*)kp0;
    v16h kb1 = *(const v16h*)(kp0 + 16 * HDc);
    v8f zero = {};
    v8f g0 = __builtin_amdgcn_wmma_f32_16x16x32_f16(false, qa16, false, kb0,
                                                    (short)0, zero, false, false);
    v8f g1 = __builtin_amdgcn_wmma_f32_16x16x32_f16(false, qa16, false, kb1,
                                                    (short)0, zero, false, false);

    // Transpose Gram from D-layout to A-layout via per-wave LDS scratch.
#pragma unroll
    for (int r = 0; r < 8; ++r) {
      gs[(r + hi * 8) * 40 + lm]      = g0[r];
      gs[(r + hi * 8) * 40 + 16 + lm] = g1[r];
    }
    float gt[16];
    {
      const float* gp = gs + lm * 40 + ko;
#pragma unroll
      for (int e = 0; e < 8; ++e)  gt[e] = gp[e];
#pragma unroll
      for (int e = 8; e < 16; ++e) gt[e] = gp[e];
    }

    // Per-pair distance + one-pass softmax weights (A-layout: lane=row lm).
    v16h wv;
#pragma unroll
    for (int e = 0; e < 16; ++e) {
      const int kidx = ko + e;   // j within chunk
      const float* kap = skA + kidx * 16;
      const float* kwp = skW + kidx * 16;
      float acc = fmaf(m2lam, gt[e], lqn) + skn[kidx];
#pragma unroll
      for (int rp = 0; rp < 16; ++rp) {
        float t  = fast_tanh(qwb[rp] + kwp[rp]);
        float ad = qa_r[rp] - kap[rp];
        float ta = t * ad;
        acc += ta * ta;
      }
      acc = fminf(fmaxf(acc, 0.0f) * INV_SQRT_HD, 50.0f);
      float w = __builtin_amdgcn_exp2f(-acc * LOG2E);
      rsum += w;
      wv[e] = (_Float16)w;
    }

    // attn @ V: weights already in A-layout; V transposed so B is contiguous.
    const _Float16* vp0 = Vt + ((size_t)bh * HDc + lm) * Sc + j0 + hi * 16;
    v16h vb0 = *(const v16h*)vp0;
    v16h vb1 = *(const v16h*)(vp0 + 16 * Sc);
    o0 = __builtin_amdgcn_wmma_f32_16x16x32_f16(false, wv, false, vb0,
                                                (short)0, o0, false, false);
    o1 = __builtin_amdgcn_wmma_f32_16x16x32_f16(false, wv, false, vb1,
                                                (short)0, o1, false, false);
  }

  // Row sums: lane l holds the partial for row l&15 over its K half.
  rsum += __shfl_xor(rsum, 16);

#pragma unroll
  for (int r = 0; r < 8; ++r) {
    int m = r + hi * 8;
    float rs  = __shfl(rsum, m);
    float inv = __builtin_amdgcn_rcpf(rs);
    size_t orow = ((size_t)b * Sc + i0 + m) * DIMc + h * HDc;
    Obuf[orow + lm]      = o0[r] * inv;
    Obuf[orow + 16 + lm] = o1[r] * inv;
  }
}

// ---------------------------------------------------------------------------
// Kernel 3: out = Obuf @ Wo.T + bo
// ---------------------------------------------------------------------------
__global__ __launch_bounds__(128) void outproj_kernel(
    const float* __restrict__ Obuf, const float* __restrict__ Wo,
    const float* __restrict__ bo, float* __restrict__ out)
{
  __shared__ float xs[8][DIMc];
  const int tid = threadIdx.x;
  const int t0  = blockIdx.x * 8;
  for (int idx = tid; idx < 8 * DIMc; idx += 128)
    xs[idx >> 7][idx & 127] = Obuf[(size_t)t0 * DIMc + idx];
  __syncthreads();

  const int d = tid;
  float acc[8];
#pragma unroll
  for (int t = 0; t < 8; ++t) acc[t] = bo[d];
  for (int c = 0; c < DIMc; ++c) {
    float w = Wo[d * DIMc + c];
#pragma unroll
    for (int t = 0; t < 8; ++t) acc[t] += xs[t][c] * w;
  }
#pragma unroll
  for (int t = 0; t < 8; ++t) out[(size_t)(t0 + t) * DIMc + d] = acc[t];
}

// ---------------------------------------------------------------------------
extern "C" void kernel_launch(void* const* d_in, const int* in_sizes, int n_in,
                              void* d_out, int out_size, void* d_ws, size_t ws_size,
                              hipStream_t stream)
{
  const float* x  = (const float*)d_in[0];
  const float* Wq = (const float*)d_in[1];
  const float* bq = (const float*)d_in[2];
  const float* Wk = (const float*)d_in[3];
  const float* bk = (const float*)d_in[4];
  const float* Wv = (const float*)d_in[5];
  const float* bv = (const float*)d_in[6];
  const float* Wo = (const float*)d_in[7];
  const float* bo = (const float*)d_in[8];
  const float* Am = (const float*)d_in[9];
  const float* ll = (const float*)d_in[10];
  const float* Wm = (const float*)d_in[11];
  const float* bm = (const float*)d_in[12];

  char*  ws  = (char*)d_ws;
  size_t off = 0;
  auto alloc = [&](size_t bytes) -> char* {
    char* p = ws + off;
    off = (off + bytes + 255) & ~(size_t)255;
    return p;
  };

  const size_t BHS = (size_t)Bc * Hc * Sc;               // 8192
  _Float16* Qh  = (_Float16*)alloc(BHS * HDc * 2);       // 512 KB
  _Float16* Kh  = (_Float16*)alloc(BHS * HDc * 2);
  _Float16* Vt  = (_Float16*)alloc(BHS * HDc * 2);
  float* qAb = (float*)alloc(BHS * RKc * 4);             // 512 KB each
  float* kAb = (float*)alloc(BHS * RKc * 4);
  float* qWb = (float*)alloc(BHS * RKc * 4);
  float* kWb = (float*)alloc(BHS * RKc * 4);
  float* qnb = (float*)alloc(BHS * 4);
  float* knb = (float*)alloc(BHS * 4);
  float* Ob  = (float*)alloc((size_t)Bc * Sc * DIMc * 4);  // 1 MB
  (void)ws_size; (void)in_sizes; (void)n_in; (void)out_size;

  proj_kernel<<<(Bc * Sc) / 8, 128, 0, stream>>>(
      x, Wq, bq, Wk, bk, Wv, bv, Am, Wm,
      Qh, Kh, Vt, qAb, kAb, qWb, kWb, qnb, knb);

  attn_kernel<<<Bc * Hc * (Sc / 64), 128, 0, stream>>>(
      Qh, Kh, Vt, qAb, kAb, qWb, kWb, qnb, knb, ll, bm, Ob);

  outproj_kernel<<<(Bc * Sc) / 8, 128, 0, stream>>>(Ob, Wo, bo, (float*)d_out);
}
